// get_adj_mx_67594195305196
// MI455X (gfx1250) — matrix operand order repaired and checked by reading, not compiled, not run on hardware
//
#include <hip/hip_runtime.h>
#include <hip/hip_fp16.h>

// Problem constants (from reference): B=64, N=1024, D=64, K=8
#define B_ 64
#define N_ 1024
#define D_ 64

typedef _Float16 v8h  __attribute__((ext_vector_type(8)));
typedef _Float16 v16h __attribute__((ext_vector_type(16)));
typedef float    v8f  __attribute__((ext_vector_type(8)));

union AV { v16h v; v8h h[2]; };

// ---------------------------------------------------------------------------
// Kernel 1: q = x@wq^T + bq ; k = x@wk^T + bk  -> stored as f16 [B*N, 64]
// 256 threads/block, 32 rows/block. Weights cached in LDS with stride-65
// padding (bank-conflict free: bank = (d*65+j)%64 = (d+j)%64, distinct/lane).
// ---------------------------------------------------------------------------
__global__ __launch_bounds__(256)
void proj_qk_f16_kernel(const float* __restrict__ x,
                        const float* __restrict__ wq, const float* __restrict__ bq,
                        const float* __restrict__ wk, const float* __restrict__ bk,
                        _Float16* __restrict__ qf, _Float16* __restrict__ kf)
{
    __shared__ float wsh[2][64 * 65];
    __shared__ float bsh[128];
    __shared__ float xs[128];          // two x rows

    const int t = threadIdx.x;
    for (int idx = t; idx < 2 * 64 * 64; idx += 256) {
        const int wsel = idx >> 12;            // 0=q weights, 1=k weights
        const int rem  = idx & 4095;
        const int d    = rem >> 6;
        const int j    = rem & 63;
        wsh[wsel][d * 65 + j] = (wsel ? wk : wq)[rem];
    }
    if (t < 64) { bsh[t] = bq[t]; bsh[64 + t] = bk[t]; }
    __syncthreads();

    const int base = blockIdx.x * 32;          // first global row of this block
    const int rs   = t >> 7;                   // which of the 2 staged rows
    const int wsel = (t >> 6) & 1;             // 0 -> q, 1 -> k
    const int d    = t & 63;
    const float* wrow = &wsh[wsel][d * 65];
    const float  bias = bsh[wsel * 64 + d];
    _Float16* dst_base = wsel ? kf : qf;

    for (int it = 0; it < 16; ++it) {
        const int r0 = base + it * 2;
        if (t < 128) xs[t] = x[(size_t)r0 * D_ + t];   // rows r0, r0+1
        __syncthreads();
        const float* xr = &xs[rs * 64];
        float acc = bias;
        #pragma unroll
        for (int j = 0; j < 64; ++j) acc = fmaf(xr[j], wrow[j], acc);
        dst_base[(size_t)(r0 + rs) * D_ + d] = (_Float16)acc;
        __syncthreads();
    }
}

// ---------------------------------------------------------------------------
// Kernel 2: per (batch, 16-row tile): scores = q . k^T via WMMA f16->f32,
// per-row top-8 / bottom-8, then stream zero rows + patch +-1 entries.
// 512 threads = 16 waves. Wave w computes column tiles [4w, 4w+4), then owns
// result row w. LDS: 16 x 1024 f32 = 64 KB (WGP has 320 KB).
// ---------------------------------------------------------------------------
__global__ __launch_bounds__(512)
void scores_topk_mask_kernel(const _Float16* __restrict__ qf,
                             const _Float16* __restrict__ kf,
                             float* __restrict__ out)
{
    __shared__ float sc[16][N_];

    const int tid  = threadIdx.x;
    const int lane = tid & 31;          // gfx1250 is wave32
    const int w    = tid >> 5;          // wave id 0..15
    const int tile = blockIdx.x;        // row tile 0..63
    const int b    = blockIdx.y;        // batch 0..63

    // ---------------- Phase A: WMMA score strip [16 x 1024] ----------------
    // A operand (16x32 f16): lane l holds row (l&15); K-halves per lane>>4.
    const int r  = lane & 15;
    const int kb = (lane >> 4) * 8;     // half-element offset inside 32-K chunk
    const _Float16* qrow = qf + ((size_t)b * N_ + tile * 16 + r) * D_;
    AV a0, a1;
    a0.h[0] = *(const v8h*)(qrow +       kb);   // K = kb..kb+7
    a0.h[1] = *(const v8h*)(qrow + 16 +  kb);   // K = 16+kb..
    a1.h[0] = *(const v8h*)(qrow + 32 +  kb);   // dims 32..63
    a1.h[1] = *(const v8h*)(qrow + 48 +  kb);

    // B operand (32x16 f16): lane l holds column (l&15), same K split as A.
    const int n   = lane & 15;
    const int kb2 = (lane >> 4) * 16;
    const int hi  = lane >> 4;

    #pragma unroll
    for (int j = 0; j < 4; ++j) {
        const int ct = w * 4 + j;                         // column tile
        const _Float16* krow = kf + ((size_t)b * N_ + ct * 16 + n) * D_;
        AV bb0, bb1;
        bb0.h[0] = *(const v8h*)(krow + kb);              // dims kb..kb+7
        bb0.h[1] = *(const v8h*)(krow + 16 + kb);         // dims 16+kb..
        bb1.h[0] = *(const v8h*)(krow + 32 + kb);         // dims 32+kb..
        bb1.h[1] = *(const v8h*)(krow + 48 + kb);
        v16h b0 = bb0.v;
        v16h b1 = bb1.v;
        v8f acc = {0.f, 0.f, 0.f, 0.f, 0.f, 0.f, 0.f, 0.f};
        acc = __builtin_amdgcn_wmma_f32_16x16x32_f16(false, a0.v, false, b0,
                                                     (short)0, acc, false, false);
        acc = __builtin_amdgcn_wmma_f32_16x16x32_f16(false, a1.v, false, b1,
                                                     (short)0, acc, false, false);
        // C/D layout: VGPR m -> row m + 8*(lane>>4); column = lane&15
        #pragma unroll
        for (int m = 0; m < 8; ++m)
            sc[m + 8 * hi][ct * 16 + n] = acc[m];
    }
    __syncthreads();

    // ---------------- Phase B: top-8 / bottom-8 for row w -------------------
    // (tanh(x/8) is monotonic -> raw scores give identical indices)
    float vv[32];
    #pragma unroll
    for (int i = 0; i < 32; ++i) vv[i] = sc[w][lane + 32 * i];

    unsigned usedMax = 0u, usedMin = 0u;
    int myTop = 0, myBot = 0;

    for (int t = 0; t < 8; ++t) {              // argmax rounds
        float bv = -3.4e38f; int gi = 0x7fffffff;
        #pragma unroll
        for (int i = 0; i < 32; ++i) {
            const bool ok = ((usedMax >> i) & 1u) == 0u;
            if (ok && vv[i] > bv) { bv = vv[i]; gi = lane + 32 * i; }
        }
        #pragma unroll
        for (int off = 16; off >= 1; off >>= 1) {
            const float ov = __shfl_xor(bv, off, 32);
            const int   oi = __shfl_xor(gi, off, 32);
            if (ov > bv || (ov == bv && oi < gi)) { bv = ov; gi = oi; }
        }
        if ((gi & 31) == lane) usedMax |= 1u << (gi >> 5);
        if (lane == t) myTop = gi;             // lanes 0..7 hold top indices
    }
    for (int t = 0; t < 8; ++t) {              // argmin rounds
        float bv = 3.4e38f; int gi = 0x7fffffff;
        #pragma unroll
        for (int i = 0; i < 32; ++i) {
            const bool ok = ((usedMin >> i) & 1u) == 0u;
            if (ok && vv[i] < bv) { bv = vv[i]; gi = lane + 32 * i; }
        }
        #pragma unroll
        for (int off = 16; off >= 1; off >>= 1) {
            const float ov = __shfl_xor(bv, off, 32);
            const int   oi = __shfl_xor(gi, off, 32);
            if (ov < bv || (ov == bv && oi < gi)) { bv = ov; gi = oi; }
        }
        if ((gi & 31) == lane) usedMin |= 1u << (gi >> 5);
        if (lane == t + 8) myBot = gi;         // lanes 8..15 hold bottom indices
    }

    // ---------------- Phase C: stream zeros + patch +-1 ---------------------
    const size_t NN      = (size_t)N_ * N_;
    const size_t rowBase = ((size_t)b * N_ + tile * 16 + w) * (size_t)N_;
    float* aff = out + rowBase;
    float* pen = out + (size_t)B_ * NN + rowBase;
    const float4 z4 = make_float4(0.f, 0.f, 0.f, 0.f);
    #pragma unroll
    for (int g = 0; g < 8; ++g) {
        *(float4*)(aff + g * 128 + lane * 4) = z4;
        *(float4*)(pen + g * 128 + lane * 4) = z4;
    }
    // guarantee zero-fill stores retire before the scatter patches
    asm volatile("s_wait_storecnt 0x0" ::: "memory");
    if (lane < 8)       aff[myTop] =  1.0f;
    else if (lane < 16) pen[myBot] = -1.0f;
}

// ---------------------------------------------------------------------------
extern "C" void kernel_launch(void* const* d_in, const int* in_sizes, int n_in,
                              void* d_out, int out_size, void* d_ws, size_t ws_size,
                              hipStream_t stream)
{
    (void)in_sizes; (void)n_in; (void)out_size; (void)ws_size;
    const float* x  = (const float*)d_in[0];
    const float* wq = (const float*)d_in[1];
    const float* bq = (const float*)d_in[2];
    const float* wk = (const float*)d_in[3];
    const float* bk = (const float*)d_in[4];
    float* out = (float*)d_out;

    // workspace: q,k as f16 [B*N, 64]  -> 2 * 64*1024*64 * 2B = 16 MB
    _Float16* qf = (_Float16*)d_ws;
    _Float16* kf = qf + (size_t)B_ * N_ * D_;

    proj_qk_f16_kernel<<<(B_ * N_) / 32, 256, 0, stream>>>(x, wq, bq, wk, bk, qf, kf);
    scores_topk_mask_kernel<<<dim3(N_ / 16, B_), 512, 0, stream>>>(qf, kf, out);
}
